// RWKV_RNN_89232240542207
// MI455X (gfx1250) — compile-verified, matrix operand order checked by reading, not run on hardware
//
#include <hip/hip_runtime.h>
#include <hip/hip_bf16.h>

typedef __bf16 bf16;
typedef __attribute__((ext_vector_type(16))) __bf16 v16bf;
typedef __attribute__((ext_vector_type(8)))  float  v8f;

#define T_ 256
#define V_ 16384
#define E_ 2048
#define H_ 32
#define S_ 64
#define L_ 4
#define F_ 7168

// ---------------------------------------------------------------- embedding
__global__ __launch_bounds__(256)
void embed_kernel(const int* __restrict__ tokens,
                  const float* __restrict__ emb,
                  float* __restrict__ x) {
    int t = blockIdx.x;
    int tok = tokens[t];
    const float* src = emb + (size_t)tok * E_;
    float* dst = x + (size_t)t * E_;
    for (int e = threadIdx.x; e < E_; e += blockDim.x) dst[e] = src[e];
}

// ------------------------------------------------------------ layernorm row
// xx = LN(x)*w + b  (biased variance, eps 1e-5). Optional bf16 copy.
__global__ __launch_bounds__(256)
void ln_kernel(const float* __restrict__ x,
               const float* __restrict__ w,
               const float* __restrict__ b,
               float* __restrict__ xx,
               bf16* __restrict__ xx_bf) {
    int t = blockIdx.x;
    const float* row = x + (size_t)t * E_;
    __shared__ float s1[256], s2[256];
    float sum = 0.f, sq = 0.f;
    for (int e = threadIdx.x; e < E_; e += 256) {
        float v = row[e];
        sum += v; sq += v * v;
    }
    s1[threadIdx.x] = sum; s2[threadIdx.x] = sq;
    __syncthreads();
    for (int st = 128; st > 0; st >>= 1) {
        if (threadIdx.x < st) {
            s1[threadIdx.x] += s1[threadIdx.x + st];
            s2[threadIdx.x] += s2[threadIdx.x + st];
        }
        __syncthreads();
    }
    float m = s1[0] * (1.f / E_);
    float var = s2[0] * (1.f / E_) - m * m;
    float rstd = rsqrtf(var + 1e-5f);
    for (int e = threadIdx.x; e < E_; e += 256) {
        float v = (row[e] - m) * rstd * w[e] + b[e];
        xx[(size_t)t * E_ + e] = v;
        if (xx_bf) xx_bf[(size_t)t * E_ + e] = (bf16)v;
    }
}

// ------------------------------------------------- token-shift mix (2 or 3)
// o = xx*tm + sx*(1-tm) = sx + tm*(xx-sx), sx = xx[t-1] (0 at t==0)
__global__ __launch_bounds__(256)
void mix_kernel(const float* __restrict__ xx,
                const float* __restrict__ tmA,
                const float* __restrict__ tmB,
                const float* __restrict__ tmC,
                bf16* __restrict__ oA,
                bf16* __restrict__ oB,
                bf16* __restrict__ oC) {
    int idx = blockIdx.x * 256 + threadIdx.x;
    if (idx >= T_ * E_) return;
    int e = idx & (E_ - 1);
    float cur = xx[idx];
    float prev = (idx >= E_) ? xx[idx - E_] : 0.f;
    float d = cur - prev;
    oA[idx] = (bf16)fmaf(tmA[e], d, prev);
    if (oB) oB[idx] = (bf16)fmaf(tmB[e], d, prev);
    if (oC) oC[idx] = (bf16)fmaf(tmC[e], d, prev);
}

// ----------------------------------------------------------- WMMA bf16 GEMM
// C[M,N] = epi( scale * A[M,K](bf16) @ W[N,K](f32)^T )
// One wave per block, 64(M) x 16(N) tile, 4 accumulators share one B frag.
// K-loop is ping-pong double buffered in steps of 64 (K % 64 == 0) so the
// next phase's global loads are in flight during the current phase's WMMAs.
// EPI: 0 = store f32; 1 = relu^2 -> bf16; 2 = resid + val -> f32;
//      3 = resid + gate*val -> f32; 4 = sigmoid(val) -> f32
template <int EPI>
__global__ __launch_bounds__(32)
void gemm_kernel(const bf16* __restrict__ A, const float* __restrict__ W,
                 void* __restrict__ C,
                 const float* __restrict__ resid, const float* __restrict__ gate,
                 int M, int N, int K, float scale) {
    int m0 = blockIdx.x * 64;
    int n0 = blockIdx.y * 16;
    int lane = threadIdx.x;
    int lr = lane & 15;
    int koff = (lane & 16) ? 8 : 0;

    const bf16*  arow = A + (size_t)(m0 + lr) * K + koff;
    const float* wrow = W + (size_t)(n0 + lr) * K + koff;

    v8f acc[4] = {};

    float4 wA[4], wB[4];
    uint4  aA[4][2], aB[4][2];

    auto loadW = [&](float4* wd, int k) {
        const float* wp = wrow + k;
        wd[0] = *(const float4*)(wp);
        wd[1] = *(const float4*)(wp + 4);
        wd[2] = *(const float4*)(wp + 16);
        wd[3] = *(const float4*)(wp + 20);
    };
    auto loadA = [&](uint4 (*ad)[2], int k) {
        #pragma unroll
        for (int mt = 0; mt < 4; ++mt) {
            const bf16* ap = arow + (size_t)mt * 16 * K + k;
            ad[mt][0] = *(const uint4*)(ap);
            ad[mt][1] = *(const uint4*)(ap + 16);
        }
    };
    auto cvtW = [&](const float4* wd) {
        v16bf b;
        #pragma unroll
        for (int q = 0; q < 4; ++q) {
            b[4 * q + 0] = (bf16)wd[q].x; b[4 * q + 1] = (bf16)wd[q].y;
            b[4 * q + 2] = (bf16)wd[q].z; b[4 * q + 3] = (bf16)wd[q].w;
        }
        return b;
    };
    auto mma = [&](const uint4 (*ad)[2], v16bf b) {
        #pragma unroll
        for (int mt = 0; mt < 4; ++mt) {
            union { v16bf v; uint4 u[2]; } af;
            af.u[0] = ad[mt][0]; af.u[1] = ad[mt][1];
            acc[mt] = __builtin_amdgcn_wmma_f32_16x16x32_bf16(
                false, af.v, false, b, (short)0, acc[mt], false, false);
        }
    };

    loadW(wA, 0);
    loadA(aA, 0);
    for (int k0 = 0; k0 < K; k0 += 64) {
        __builtin_prefetch(wrow + k0 + 512, 0, 0);   // weight stream, ~2KB ahead
        loadW(wB, k0 + 32);
        loadA(aB, k0 + 32);
        mma(aA, cvtW(wA));
        if (k0 + 64 < K) { loadW(wA, k0 + 64); loadA(aA, k0 + 64); }
        mma(aB, cvtW(wB));
    }

    // ---- epilogue. C/D layout: VGPR g, lane<16 -> M=g, lane>=16 -> M=g+8
    float* Cf = (float*)C;
    bf16*  Cb = (bf16*)C;
    int n = n0 + lr;
    int mb = m0 + ((lane & 16) ? 8 : 0);
    #pragma unroll
    for (int mt = 0; mt < 4; ++mt) {
        #pragma unroll
        for (int g = 0; g < 8; ++g) {
            int m = mb + mt * 16 + g;
            size_t idx = (size_t)m * N + n;
            float val = acc[mt][g] * scale;
            if constexpr (EPI == 0) {
                Cf[idx] = val;
            } else if constexpr (EPI == 1) {
                float r = fmaxf(val, 0.f);
                Cb[idx] = (bf16)(r * r);
            } else if constexpr (EPI == 2) {
                Cf[idx] = resid[idx] + val;
            } else if constexpr (EPI == 3) {
                Cf[idx] = fmaf(gate[idx], val, resid[idx]);
            } else {
                Cf[idx] = 1.f / (1.f + __expf(-val));
            }
        }
    }
}

// -------------------------------------------------------------- WKV-v5 scan
// per head: st[i][j]; wkv[t,j] = sum_i r[t,i]*(tf[i]*k[t,i]*v[t,j] + st[i][j])
//           st[i][j] = td[i]*st[i][j] + k[t,i]*v[t,j]
// block = 256 threads = 4 i-chunks(16) x 64 j
__global__ __launch_bounds__(256)
void wkv_kernel(const float* __restrict__ r, const float* __restrict__ k,
                const float* __restrict__ v,
                const float* __restrict__ tf, const float* __restrict__ td,
                float* __restrict__ a) {
    int h  = blockIdx.x;
    int j  = threadIdx.x & 63;
    int ic = threadIdx.x >> 6;           // 0..3
    __shared__ float ksh[64], rsh[64], part[4][64];
    float st[16], tfr[16], tdr[16];
    #pragma unroll
    for (int ii = 0; ii < 16; ++ii) {
        st[ii] = 0.f;
        int i = ic * 16 + ii;
        tfr[ii] = tf[h * S_ + i];
        tdr[ii] = td[h * S_ + i];
    }
    for (int t = 0; t < T_; ++t) {
        size_t base = (size_t)t * E_ + h * S_;
        if (threadIdx.x < 64) { ksh[j] = k[base + j]; rsh[j] = r[base + j]; }
        __syncthreads();
        float vj = v[base + j];
        float acc = 0.f;
        #pragma unroll
        for (int ii = 0; ii < 16; ++ii) {
            int i = ic * 16 + ii;
            float kv = ksh[i] * vj;
            acc = fmaf(rsh[i], fmaf(tfr[ii], kv, st[ii]), acc);
            st[ii] = fmaf(tdr[ii], st[ii], kv);
        }
        part[ic][j] = acc;
        __syncthreads();
        if (ic == 0)
            a[base + j] = part[0][j] + part[1][j] + part[2][j] + part[3][j];
        // next iteration's first __syncthreads orders part-read vs next write
    }
}

// ------------------------------------------ per-(t,h) instance-norm + affine
__global__ __launch_bounds__(64)
void inorm_kernel(const float* __restrict__ a,
                  const float* __restrict__ w, const float* __restrict__ b,
                  bf16* __restrict__ out) {
    int th = blockIdx.x;
    int t = th / H_, h = th % H_;
    int j = threadIdx.x;
    const float* row = a + ((size_t)t * H_ + h) * S_;
    __shared__ float s1[64], s2[64];
    float v = row[j];
    s1[j] = v; s2[j] = v * v;
    __syncthreads();
    for (int st = 32; st > 0; st >>= 1) {
        if (j < st) { s1[j] += s1[j + st]; s2[j] += s2[j + st]; }
        __syncthreads();
    }
    float m = s1[0] * (1.f / S_);
    float var = s2[0] * (1.f / S_) - m * m;
    float rstd = rsqrtf(var + 1e-5f);
    int e = h * S_ + j;
    out[(size_t)t * E_ + e] = (bf16)(((v - m) * rstd) * w[e] + b[e]);
}

// ===========================================================================
extern "C" void kernel_launch(void* const* d_in, const int* in_sizes, int n_in,
                              void* d_out, int out_size, void* d_ws, size_t ws_size,
                              hipStream_t stream) {
    const int*   tokens   = (const int*)d_in[0];
    const float* emb      = (const float*)d_in[1];
    const float* ln1_w    = (const float*)d_in[2];
    const float* ln1_b    = (const float*)d_in[3];
    const float* tmk      = (const float*)d_in[4];
    const float* tmv      = (const float*)d_in[5];
    const float* tmr      = (const float*)d_in[6];
    const float* tfirst   = (const float*)d_in[7];
    const float* tdecay   = (const float*)d_in[8];
    const float* att_kw   = (const float*)d_in[9];
    const float* att_vw   = (const float*)d_in[10];
    const float* att_rw   = (const float*)d_in[11];
    const float* att_ow   = (const float*)d_in[12];
    const float* lnx_w    = (const float*)d_in[13];
    const float* lnx_b    = (const float*)d_in[14];
    const float* ln2_w    = (const float*)d_in[15];
    const float* ln2_b    = (const float*)d_in[16];
    const float* ffn_tmk  = (const float*)d_in[17];
    const float* ffn_tmr  = (const float*)d_in[18];
    const float* ffn_kw   = (const float*)d_in[19];
    const float* ffn_vw   = (const float*)d_in[20];
    const float* ffn_rw   = (const float*)d_in[21];
    const float* lnout_w  = (const float*)d_in[22];
    const float* lnout_b  = (const float*)d_in[23];
    const float* head_w   = (const float*)d_in[24];

    // ---------------- workspace carve-up (256B aligned)
    char* ws = (char*)d_ws;
    size_t off = 0;
    auto carve = [&](size_t bytes) -> void* {
        void* p = ws + off;
        off += (bytes + 255) & ~(size_t)255;
        return p;
    };
    float* x    = (float*)carve((size_t)T_ * E_ * 4);
    float* xx   = (float*)carve((size_t)T_ * E_ * 4);
    bf16*  xkb  = (bf16*)carve((size_t)T_ * E_ * 2);
    bf16*  xvb  = (bf16*)carve((size_t)T_ * E_ * 2);
    bf16*  xrb  = (bf16*)carve((size_t)T_ * E_ * 2);
    float* rbuf = (float*)carve((size_t)T_ * E_ * 4);
    float* kbuf = (float*)carve((size_t)T_ * E_ * 4);
    float* vbuf = (float*)carve((size_t)T_ * E_ * 4);
    float* abuf = (float*)carve((size_t)T_ * E_ * 4);
    bf16*  abf  = (bf16*)carve((size_t)T_ * E_ * 2);
    float* rr   = (float*)carve((size_t)T_ * E_ * 4);
    bf16*  kkb  = (bf16*)carve((size_t)T_ * F_ * 2);
    bf16*  xbf  = (bf16*)carve((size_t)T_ * E_ * 2);

    dim3 gE(T_ / 64, E_ / 16);   // GEMM N = E
    dim3 gF(T_ / 64, F_ / 16);   // GEMM N = F
    dim3 gV(T_ / 64, V_ / 16);   // GEMM N = V

    embed_kernel<<<T_, 256, 0, stream>>>(tokens, emb, x);

    for (int l = 0; l < L_; ++l) {
        size_t oE  = (size_t)l * E_;
        size_t oEE = (size_t)l * E_ * E_;
        size_t oFE = (size_t)l * (size_t)F_ * E_;
        size_t oHS = (size_t)l * H_ * S_;

        // ---- time mixing
        ln_kernel<<<T_, 256, 0, stream>>>(x, ln1_w + oE, ln1_b + oE, xx, nullptr);
        mix_kernel<<<(T_ * E_) / 256, 256, 0, stream>>>(
            xx, tmk + oE, tmv + oE, tmr + oE, xkb, xvb, xrb);
        gemm_kernel<0><<<gE, 32, 0, stream>>>(xrb, att_rw + oEE, rbuf,
                                              nullptr, nullptr, T_, E_, E_, 1.f);
        gemm_kernel<0><<<gE, 32, 0, stream>>>(xkb, att_kw + oEE, kbuf,
                                              nullptr, nullptr, T_, E_, E_, 1.f);
        gemm_kernel<0><<<gE, 32, 0, stream>>>(xvb, att_vw + oEE, vbuf,
                                              nullptr, nullptr, T_, E_, E_, 1.f / 32.f);
        wkv_kernel<<<H_, 256, 0, stream>>>(rbuf, kbuf, vbuf,
                                           tfirst + oHS, tdecay + oHS, abuf);
        inorm_kernel<<<T_ * H_, 64, 0, stream>>>(abuf, lnx_w + oE, lnx_b + oE, abf);
        gemm_kernel<2><<<gE, 32, 0, stream>>>(abf, att_ow + oEE, x,
                                              x, nullptr, T_, E_, E_, 1.f);

        // ---- channel mixing
        ln_kernel<<<T_, 256, 0, stream>>>(x, ln2_w + oE, ln2_b + oE, xx, nullptr);
        mix_kernel<<<(T_ * E_) / 256, 256, 0, stream>>>(
            xx, ffn_tmk + oE, ffn_tmr + oE, nullptr, xkb, xrb, nullptr);
        gemm_kernel<4><<<gE, 32, 0, stream>>>(xrb, ffn_rw + oEE, rr,
                                              nullptr, nullptr, T_, E_, E_, 1.f);
        gemm_kernel<1><<<gF, 32, 0, stream>>>(xkb, ffn_kw + oFE, kkb,
                                              nullptr, nullptr, T_, F_, E_, 1.f);
        gemm_kernel<3><<<gE, 32, 0, stream>>>(kkb, ffn_vw + oFE, x,
                                              x, rr, T_, E_, F_, 1.f);
    }

    // ---- head
    ln_kernel<<<T_, 256, 0, stream>>>(x, lnout_w, lnout_b, xx, xbf);
    gemm_kernel<0><<<gV, 32, 0, stream>>>(xbf, head_w, d_out,
                                          nullptr, nullptr, T_, V_, E_, 1.f);
}
